// MultiHeadAttention_80178449482241
// MI455X (gfx1250) — compile-verified, matrix operand order checked by reading, not run on hardware
//
#include <hip/hip_runtime.h>
#include <hip/hip_bf16.h>
#include <stdint.h>

#ifndef __has_builtin
#define __has_builtin(x) 0
#endif
#if __has_builtin(__builtin_amdgcn_tensor_load_to_lds) && \
    __has_builtin(__builtin_amdgcn_s_wait_tensorcnt)
#define HAVE_TDM 1
#else
#define HAVE_TDM 0
#endif

// ---------------------------------------------------------------------------
// Types for CDNA5 WMMA (wave32): v_wmma_f32_16x16x32_bf16
// ---------------------------------------------------------------------------
typedef __bf16 bf16_t;
typedef bf16_t v16bf __attribute__((ext_vector_type(16)));
typedef bf16_t v8bf  __attribute__((ext_vector_type(8)));
typedef bf16_t v2bf  __attribute__((ext_vector_type(2)));
typedef float  v8f   __attribute__((ext_vector_type(8)));
typedef unsigned int u32x4 __attribute__((ext_vector_type(4)));
typedef int          i32x8 __attribute__((ext_vector_type(8)));
typedef int          i32x4 __attribute__((ext_vector_type(4)));

__device__ __forceinline__ v8f wmma_bf16(v16bf a, v16bf b, v8f c) {
    // (neg_a, A, neg_b, B, c_mod, C, reuse_a, reuse_b)
    return __builtin_amdgcn_wmma_f32_16x16x32_bf16(false, a, false, b,
                                                   (short)0, c, false, false);
}

// Load a 16x32 bf16 A/B fragment from a (LDS or global) row.
// Lane L holds M=L%16; kb=(L/16)*8; VGPRs 0-3 = K kb..kb+7, 4-7 = K 16+kb..+7.
__device__ __forceinline__ v16bf load_frag(const bf16_t* row, int kb) {
    v8bf lo = *(const v8bf*)(row + kb);
    v8bf hi = *(const v8bf*)(row + 16 + kb);
    v16bf r;
#pragma unroll
    for (int i = 0; i < 8; ++i) { r[i] = lo[i]; r[8 + i] = hi[i]; }
    return r;
}

// ---------------------------------------------------------------------------
// Generic GEMM: C[M,N] = alpha * A(MxK) @ Bm(KxN) + bias[n]
//   MODE 0: Bm[k][n] = B[k*N + n]                     (row-major KxN)
//   MODE 1: Bm[k][n] = B[n*K + k]                     (x @ W.T, W is NxK)
//   MODE 2: Bm[k][n] = B[(n>>6)*K*64 + k*64 + (n&63)] (per-head (H,D,64) pack)
// ABF / CBF (compile-time) select bf16 A input / bf16 C output — templated so
// the staging loops are branch-free straight-line clauses.
// Tiles 128x128x32, double-buffered LDS, 8 waves (2M x 4N), 64x32 per wave.
// ---------------------------------------------------------------------------
#define TM 128
#define TN 128
#define TK 32
#define LDP 40   // padded LDS row (bf16): 80B, keeps frag reads 16B-aligned

template <int MODE, int ABF, int CBF>
__global__ __launch_bounds__(256)
void gemm_bf16_kernel(const void* __restrict__ A_, const float* __restrict__ Bm,
                      const float* __restrict__ bias, void* __restrict__ C_,
                      int M, int N, int K, float alpha)
{
    __shared__ bf16_t As[2][TM * LDP];
    __shared__ bf16_t Bs[2][TN * LDP];   // stored transposed: Bs[n][k]

    const int tid  = threadIdx.x;
    const int wid  = tid >> 5;
    const int lane = tid & 31;
    const int l16  = lane & 15;
    const int hi8  = (lane >> 4) * 8;
    const int kb   = hi8;
    const int bm0  = blockIdx.y * TM;
    const int bn0  = blockIdx.x * TN;
    const int wm   = (wid & 1) * 64;
    const int wn   = (wid >> 1) * 32;

    auto stage = [&](int buf, int k0) {
#pragma unroll
        for (int idx = tid; idx < TM * 16; idx += 256) {
            const int r  = idx >> 4;
            const int kp = (idx & 15) << 1;
            bf16_t x0, x1;
            if (ABF) {
                const v2bf v = *(const v2bf*)((const bf16_t*)A_ +
                                              (size_t)(bm0 + r) * K + k0 + kp);
                x0 = v[0]; x1 = v[1];
            } else {
                const float2 v = *(const float2*)((const float*)A_ +
                                                  (size_t)(bm0 + r) * K + k0 + kp);
                x0 = (bf16_t)v.x; x1 = (bf16_t)v.y;
            }
            As[buf][r * LDP + kp]     = x0;
            As[buf][r * LDP + kp + 1] = x1;
        }
#pragma unroll
        for (int idx = tid; idx < TN * 16; idx += 256) {
            const int n  = idx >> 4;
            const int kp = (idx & 15) << 1;
            const int gn = bn0 + n;
            const int gk = k0 + kp;
            float b0, b1;
            if (MODE == 1) {
                const float2 v = *(const float2*)(Bm + (size_t)gn * K + gk);
                b0 = v.x; b1 = v.y;
            } else if (MODE == 2) {
                const float* p = Bm + (size_t)(gn >> 6) * K * 64 +
                                 (size_t)gk * 64 + (gn & 63);
                b0 = p[0]; b1 = p[64];
            } else {
                b0 = Bm[(size_t)gk * N + gn];
                b1 = Bm[(size_t)(gk + 1) * N + gn];
            }
            Bs[buf][n * LDP + kp]     = (bf16_t)b0;
            Bs[buf][n * LDP + kp + 1] = (bf16_t)b1;
        }
        if (k0 + TK < K)   // CDNA5 global_prefetch_b8, one tile further ahead
            __builtin_prefetch((const char*)A_ +
                ((size_t)(bm0 + (tid >> 1)) * K + k0 + TK) * (ABF ? 2 : 4), 0, 0);
    };

    v8f acc[4][2];
#pragma unroll
    for (int i = 0; i < 4; ++i)
#pragma unroll
        for (int j = 0; j < 2; ++j) acc[i][j] = (v8f)0.0f;

    stage(0, 0);
    int p = 0;
    for (int k0 = 0; k0 < K; k0 += TK) {
        __syncthreads();                       // buffer p published
        if (k0 + TK < K) stage(p ^ 1, k0 + TK);// overlap next-tile staging
        v16bf bfr[2];
#pragma unroll
        for (int fn = 0; fn < 2; ++fn)
            bfr[fn] = load_frag(&Bs[p][(wn + fn * 16 + l16) * LDP], kb);
#pragma unroll
        for (int fm = 0; fm < 4; ++fm) {
            const v16bf af = load_frag(&As[p][(wm + fm * 16 + l16) * LDP], kb);
#pragma unroll
            for (int fn = 0; fn < 2; ++fn)
                acc[fm][fn] = wmma_bf16(af, bfr[fn], acc[fm][fn]);
        }
        p ^= 1;
    }

    // epilogue: C-frag row = j + hi8, col = l16
#pragma unroll
    for (int fm = 0; fm < 4; ++fm)
#pragma unroll
        for (int fn = 0; fn < 2; ++fn) {
            const int col = bn0 + wn + fn * 16 + l16;
            const float bv = bias ? bias[col] : 0.0f;
#pragma unroll
            for (int j = 0; j < 8; ++j) {
                const int row = bm0 + wm + fm * 16 + j + hi8;
                const float v = alpha * acc[fm][fn][j] + bv;
                if (CBF) ((bf16_t*)C_)[(size_t)row * N + col] = (bf16_t)v;
                else     ((float*)C_)[(size_t)row * N + col] = v;
            }
        }
}

// ---------------------------------------------------------------------------
// LayerNorm + ReLU + residual: out = x + relu(LN(u)*g + b), one block per row
// ---------------------------------------------------------------------------
__global__ __launch_bounds__(256)
void ln_relu_res_kernel(const float* __restrict__ u, const float* __restrict__ x,
                        const float* __restrict__ g, const float* __restrict__ b,
                        float* __restrict__ out, int D)
{
    const int row = blockIdx.x;
    const float* ur = u + (size_t)row * D;
    const float* xr = x + (size_t)row * D;
    float* orow = out + (size_t)row * D;

    float s = 0.f, s2 = 0.f;
    for (int i = threadIdx.x; i < D; i += 256) {
        const float v = ur[i];
        s += v; s2 += v * v;
    }
#pragma unroll
    for (int off = 16; off > 0; off >>= 1) {
        s  += __shfl_xor(s,  off, 32);
        s2 += __shfl_xor(s2, off, 32);
    }
    __shared__ float red[2][8];
    if ((threadIdx.x & 31) == 0) {
        red[0][threadIdx.x >> 5] = s;
        red[1][threadIdx.x >> 5] = s2;
    }
    __syncthreads();
    float ts = 0.f, ts2 = 0.f;
#pragma unroll
    for (int i = 0; i < 8; ++i) { ts += red[0][i]; ts2 += red[1][i]; }
    const float mu  = ts / (float)D;
    const float var = ts2 / (float)D - mu * mu;
    const float rs  = rsqrtf(var + 1e-5f);
    for (int i = threadIdx.x; i < D; i += 256) {
        const float y = (ur[i] - mu) * rs * g[i] + b[i];
        orow[i] = xr[i] + fmaxf(y, 0.f);
    }
}

// ---------------------------------------------------------------------------
// Flash attention: block = (head,batch) x 128 query rows; 8 waves x 16 rows.
// Q/K/V/heads are bf16 (B, N, H*64). nf folded into Q and K. Double-buffered
// K/V tiles; mask tile fetched by the Tensor Data Mover one tile ahead.
// ---------------------------------------------------------------------------
#define AK_LDP 72   // padded bf16 row stride for K/V tiles (144B)
#define P_LDP  64   // P staging row stride (128B)

__global__ __launch_bounds__(256)
void flash_attn_kernel(const bf16_t* __restrict__ Q, const bf16_t* __restrict__ Kb,
                       const bf16_t* __restrict__ Vb,
                       const unsigned char* __restrict__ mask,
                       bf16_t* __restrict__ Ob, int Nn)
{
    constexpr int DM = 512, HH = 8;
    __shared__ bf16_t Ks[2][64 * AK_LDP];       // [key][dk]
    __shared__ bf16_t Vs[2][64 * AK_LDP];       // transposed: [dv][key]
    __shared__ bf16_t Ps[8 * 16 * P_LDP];       // per-wave P tiles [row][key]
    __shared__ unsigned char Ms[128 * 64];      // mask tile [qrow][key]

    const int qt  = blockIdx.x;
    const int hb  = blockIdx.y;
    const int bat = hb / HH;
    const int hd  = hb % HH;
    const int tid = threadIdx.x, wid = tid >> 5, lane = tid & 31;
    const int l16 = lane & 15, hi8 = (lane >> 4) * 8, kb = hi8;
    const int q0w = qt * 128 + wid * 16;
    const size_t mbase = (size_t)bat * Nn * Nn;

    auto stage_kv = [&](int buf, int kt) {
#pragma unroll
        for (int idx = tid; idx < 64 * 8; idx += 256) {   // K rows, b128 copies
            const int r = idx >> 3, seg = (idx & 7) * 8;
            *(v8bf*)(&Ks[buf][r * AK_LDP + seg]) =
                *(const v8bf*)(Kb + (size_t)(bat * Nn + kt * 64 + r) * DM +
                               hd * 64 + seg);
        }
#pragma unroll
        for (int idx = tid; idx < 64 * 32; idx += 256) {  // V transposed
            const int r = idx >> 5, cp = (idx & 31) << 1;
            const v2bf v = *(const v2bf*)(Vb + (size_t)(bat * Nn + kt * 64 + r) * DM +
                                          hd * 64 + cp);
            Vs[buf][(cp)     * AK_LDP + r] = v[0];
            Vs[buf][(cp + 1) * AK_LDP + r] = v[1];
        }
    };

    auto stage_mask = [&](int kt) {
#if HAVE_TDM
        if (wid == 0) {
            // 2D Tensor-DMA descriptor: 1-byte elems, tile 64x128, stride N.
            const size_t moff = mbase + (size_t)(qt * 128) * Nn + (size_t)kt * 64;
            const uint64_t ga = (uint64_t)(uintptr_t)(mask + moff);
            const uint32_t la = (uint32_t)(uintptr_t)(void*)Ms;
            u32x4 g0;
            g0[0] = (unsigned)__builtin_amdgcn_readfirstlane(1);        // count=1
            g0[1] = (unsigned)__builtin_amdgcn_readfirstlane((int)la);  // lds_addr
            g0[2] = (unsigned)__builtin_amdgcn_readfirstlane((int)(uint32_t)ga);
            g0[3] = (unsigned)__builtin_amdgcn_readfirstlane(
                        (int)(((uint32_t)(ga >> 32) & 0x01FFFFFFu) | 0x80000000u));
            i32x8 g1;
            g1[0] = 0;                        // wg_mask=0, data_size=1B
            g1[1] = (int)(1024u << 16);       // tensor_dim0 = 1024 (lo16)
            g1[2] = (int)(1024u << 16);       // dim0 hi | tensor_dim1 = 1024 (lo16)
            g1[3] = (int)(64u << 16);         // dim1 hi | tile_dim0 = 64
            g1[4] = 128;                      // tile_dim1 = 128, tile_dim2 = 0
            g1[5] = 1024;                     // tensor_dim0_stride = 1024
            g1[6] = 0;
            g1[7] = 0;
            const i32x4 gz4 = {0, 0, 0, 0};
            const i32x8 gz8 = {0, 0, 0, 0, 0, 0, 0, 0};
            // clang-23 / therock-10.0 lane: 6-arg form
            __builtin_amdgcn_tensor_load_to_lds(g0, g1, gz4, gz4, gz8, 0);
        }
#else
        const size_t moff = mbase + (size_t)(qt * 128) * Nn + (size_t)kt * 64;
#pragma unroll
        for (int idx = tid; idx < 128 * 16; idx += 256) {
            const int r = idx >> 4, c4 = (idx & 15) << 2;
            *(uint32_t*)(Ms + r * 64 + c4) =
                *(const uint32_t*)(mask + moff + (size_t)r * Nn + c4);
        }
#endif
    };

    // ---- Q fragments (bf16, straight from global) ------------------------
    const bf16_t* qrow = Q + (size_t)(bat * Nn + q0w + l16) * DM + hd * 64;
    v16bf qa[2];
    qa[0] = load_frag(qrow, kb);
    qa[1] = load_frag(qrow + 32, kb);

    v8f O[4];
#pragma unroll
    for (int vf = 0; vf < 4; ++vf) O[vf] = (v8f)0.0f;
    float mrow[8], lrow[8];
#pragma unroll
    for (int j = 0; j < 8; ++j) { mrow[j] = -3.0e38f; lrow[j] = 0.f; }

    const int nkt = Nn / 64;
    stage_kv(0, 0);
    stage_mask(0);
    int p = 0;
    for (int kt = 0; kt < nkt; ++kt) {
#if HAVE_TDM
        if (wid == 0) __builtin_amdgcn_s_wait_tensorcnt(0);
#endif
        __syncthreads();                       // K/V buffer p + mask published
        if (kt + 1 < nkt) stage_kv(p ^ 1, kt + 1);

        // ---- S = Q K^T ----------------------------------------------------
        v8f s[4];
#pragma unroll
        for (int nf = 0; nf < 4; ++nf) {
            const bf16_t* krow = &Ks[p][(nf * 16 + l16) * AK_LDP];
            v8f z = (v8f)0.0f;
            z = wmma_bf16(qa[0], load_frag(krow,      kb), z);
            z = wmma_bf16(qa[1], load_frag(krow + 32, kb), z);
            s[nf] = z;
        }

        // ---- mask (from LDS tile) + online softmax ------------------------
        float rmax[8], rsum[8], corr[8];
#pragma unroll
        for (int j = 0; j < 8; ++j) { rmax[j] = -3.0e38f; rsum[j] = 0.f; }
#pragma unroll
        for (int nf = 0; nf < 4; ++nf)
#pragma unroll
            for (int j = 0; j < 8; ++j) {
                const int lr = wid * 16 + j + hi8;          // row in block tile
                float sv = s[nf][j];
                if (Ms[lr * 64 + nf * 16 + l16]) sv = -1e10f;
                s[nf][j] = sv;
                rmax[j] = fmaxf(rmax[j], sv);
            }
#pragma unroll
        for (int j = 0; j < 8; ++j) {
#pragma unroll
            for (int off = 1; off < 16; off <<= 1)
                rmax[j] = fmaxf(rmax[j], __shfl_xor(rmax[j], off, 16));
            const float mnew = fmaxf(mrow[j], rmax[j]);
            corr[j] = __expf(mrow[j] - mnew);
            mrow[j] = mnew;
        }
#pragma unroll
        for (int nf = 0; nf < 4; ++nf)
#pragma unroll
            for (int j = 0; j < 8; ++j) {
                const float pv = __expf(s[nf][j] - mrow[j]);
                s[nf][j] = pv;
                rsum[j] += pv;
            }
#pragma unroll
        for (int j = 0; j < 8; ++j) {
#pragma unroll
            for (int off = 1; off < 16; off <<= 1)
                rsum[j] += __shfl_xor(rsum[j], off, 16);
            lrow[j] = lrow[j] * corr[j] + rsum[j];
        }
#pragma unroll
        for (int vf = 0; vf < 4; ++vf)
#pragma unroll
            for (int j = 0; j < 8; ++j) O[vf][j] *= corr[j];

        // ---- P (C-layout) -> LDS bf16 -------------------------------------
        bf16_t* Pw = Ps + wid * 16 * P_LDP;
#pragma unroll
        for (int nf = 0; nf < 4; ++nf)
#pragma unroll
            for (int j = 0; j < 8; ++j)
                Pw[(j + hi8) * P_LDP + nf * 16 + l16] = (bf16_t)s[nf][j];
        __syncthreads();                       // mask consumed, P visible

        if (kt + 1 < nkt) stage_mask(kt + 1);  // TDM fetch next mask tile

        // ---- O += P @ V ---------------------------------------------------
        const bf16_t* prow = Pw + l16 * P_LDP;
        const v16bf pa0 = load_frag(prow,      kb);
        const v16bf pa1 = load_frag(prow + 32, kb);
#pragma unroll
        for (int vf = 0; vf < 4; ++vf) {
            const bf16_t* vrow = &Vs[p][(vf * 16 + l16) * AK_LDP];
            O[vf] = wmma_bf16(pa0, load_frag(vrow,      kb), O[vf]);
            O[vf] = wmma_bf16(pa1, load_frag(vrow + 32, kb), O[vf]);
        }
        p ^= 1;
    }

    // ---- normalize, write heads (bf16) -----------------------------------
#pragma unroll
    for (int vf = 0; vf < 4; ++vf)
#pragma unroll
        for (int j = 0; j < 8; ++j) {
            const int qr = q0w + j + hi8;
            Ob[(size_t)(bat * Nn + qr) * DM + hd * 64 + vf * 16 + l16] =
                (bf16_t)(O[vf][j] / lrow[j]);
        }
}

// ---------------------------------------------------------------------------
// Host orchestration
// ---------------------------------------------------------------------------
extern "C" void kernel_launch(void* const* d_in, const int* in_sizes, int n_in,
                              void* d_out, int out_size, void* d_ws, size_t ws_size,
                              hipStream_t stream) {
    (void)in_sizes; (void)n_in; (void)out_size; (void)ws_size;
    const float* q    = (const float*)d_in[0];
    const unsigned char* mask = (const unsigned char*)d_in[1];
    const float* eps1 = (const float*)d_in[2];
    const float* U_w  = (const float*)d_in[3];
    const float* U_b  = (const float*)d_in[4];
    const float* V_w  = (const float*)d_in[5];
    const float* V_b  = (const float*)d_in[6];
    const float* ln_g = (const float*)d_in[7];
    const float* ln_b = (const float*)d_in[8];
    const float* Wq   = (const float*)d_in[9];
    const float* Wk   = (const float*)d_in[10];
    const float* Wv   = (const float*)d_in[11];
    const float* Wo   = (const float*)d_in[12];
    float* out = (float*)d_out;

    const int Bb = 8, Nn = 1024, Dd = 512, Hh = 8;
    const int M = Bb * Nn;                   // 8192
    const size_t SZ = (size_t)M * Dd;        // 4 Mi elems

    float* ws   = (float*)d_ws;
    float* hbuf = ws;                        // hidden h
    float* xbuf = ws + SZ;                   // x_l
    float* tbuf = ws + 2 * SZ;               // t   -> later Q/K (bf16)
    float* ubuf = ws + 3 * SZ;               // u   -> later V/heads (bf16)
    bf16_t* Qb = (bf16_t*)tbuf;
    bf16_t* Kc = Qb + SZ;
    bf16_t* Vc = (bf16_t*)ubuf;
    bf16_t* Hd = Vc + SZ;

    const dim3 blk(256);
    const dim3 ggrid(Dd / TN, M / TM);       // (4, 64)

    // ---- 3 MLP layers (fp32 activations, bf16 WMMA inside) ---------------
    const float* hin = q;
    for (int l = 0; l < 3; ++l) {
        gemm_bf16_kernel<0, 0, 0><<<ggrid, blk, 0, stream>>>(
            hin, eps1, nullptr, xbuf, M, Dd, Dd, 1.0f);                  // x = h@eps1
        gemm_bf16_kernel<1, 0, 0><<<ggrid, blk, 0, stream>>>(
            xbuf, U_w + (size_t)l * Dd * Dd, U_b + l * Dd, tbuf,
            M, Dd, Dd, 1.0f);                                            // t = x@Uw.T+Ub
        gemm_bf16_kernel<1, 0, 0><<<ggrid, blk, 0, stream>>>(
            tbuf, V_w + (size_t)l * Dd * Dd, V_b + l * Dd, ubuf,
            M, Dd, Dd, 1.0f);                                            // u = t@Vw.T+Vb
        ln_relu_res_kernel<<<dim3(M), blk, 0, stream>>>(
            ubuf, xbuf, ln_g + l * Dd, ln_b + l * Dd, hbuf, Dd);         // h = x+relu(LN)
        hin = hbuf;
    }

    // ---- QKV projections, bf16 outputs (nf folded into Q and K) ----------
    const float nf = 0.125f;  // 1/sqrt(64)
    gemm_bf16_kernel<2, 0, 1><<<ggrid, blk, 0, stream>>>(q,    Wq, nullptr, Qb, M, Dd, Dd, nf);
    gemm_bf16_kernel<2, 0, 1><<<ggrid, blk, 0, stream>>>(hbuf, Wk, nullptr, Kc, M, Dd, Dd, nf);
    gemm_bf16_kernel<2, 0, 1><<<ggrid, blk, 0, stream>>>(hbuf, Wv, nullptr, Vc, M, Dd, Dd, 1.0f);

    // ---- flash attention -> heads (bf16) ---------------------------------
    flash_attn_kernel<<<dim3(Nn / 128, Bb * Hh), 256, 0, stream>>>(
        Qb, Kc, Vc, mask, Hd, Nn);

    // ---- output projection: W_out (H,DK,D) == row-major (512,512) --------
    gemm_bf16_kernel<0, 1, 0><<<ggrid, blk, 0, stream>>>(
        Hd, Wo, nullptr, out, M, Dd, Dd, 1.0f);
}
